// Model_86268713107834
// MI455X (gfx1250) — matrix operand order repaired and checked by reading, not compiled, not run on hardware
//
#include <hip/hip_runtime.h>

#define H       128
#define G4      512
#define TT      60
#define DF      16
#define NNODES  4000
#define WSTR    520   // 512 + 8-half pad (16B) to avoid LDS bank conflicts
#define HSTR    136   // 128 + 8-half pad
#define NTILE   (NNODES / 32)

typedef __attribute__((ext_vector_type(16))) _Float16 v16h;
typedef __attribute__((ext_vector_type(8)))  _Float16 v8h;
typedef __attribute__((ext_vector_type(8)))  float    v8f;
typedef __attribute__((__vector_size__(4 * sizeof(int)))) int v4i;

#define AS1 __attribute__((address_space(1)))
#define AS3 __attribute__((address_space(3)))

// --- CDNA5 async global->LDS copy (ASYNCcnt-tracked), builtin or asm ------
#if defined(__HIP_DEVICE_COMPILE__) && __has_builtin(__builtin_amdgcn_global_load_async_to_lds_b128)
#define ASYNC_B128(gp, lp) \
  __builtin_amdgcn_global_load_async_to_lds_b128((AS1 v4i*)(gp), (AS3 v4i*)(lp), 0, 0)
#else
#define ASYNC_B128(gp, lp) \
  asm volatile("global_load_async_to_lds_b128 %0, %1, off" \
               :: "v"((unsigned)(unsigned long long)(lp)), \
                  "v"((unsigned long long)(gp)) : "memory")
#endif

#if defined(__HIP_DEVICE_COMPILE__) && __has_builtin(__builtin_amdgcn_s_wait_asynccnt)
#define WAIT_ASYNC(n) do { __builtin_amdgcn_s_wait_asynccnt(n); asm volatile("" ::: "memory"); } while (0)
#else
#define WAIT_ASYNC(n) asm volatile("s_wait_asynccnt %0" :: "i"(n) : "memory")
#endif

__device__ __forceinline__ float sigf(float x)      { return 1.f / (1.f + __expf(-x)); }
__device__ __forceinline__ float tanh_fast(float x) { return 2.f / (1.f + __expf(-2.f * x)) - 1.f; }

__device__ __forceinline__ v16h cat16(v8h lo, v8h hi) {
  return __builtin_shufflevector(lo, hi, 0,1,2,3,4,5,6,7,8,9,10,11,12,13,14,15);
}

__device__ __forceinline__ v8f wmma_f16(v16h a, v16h b, v8f c) {
  // D = A(16x32 f16) * B(32x16 f16) + C(16x16 f32)
  return __builtin_amdgcn_wmma_f32_16x16x32_f16(false, a, false, b, (short)0, c, false, false);
}

// ---------------------------------------------------------------------------
// Fused 2-layer LSTM. One block = 16 nodes, 8 waves; wave w owns gate columns
// {g*128 + w*16 .. +16} for g=i,f,g,o, i.e. hidden columns w*16..w*16+16.
// Layer-1 weights: transposed f16 in LDS (B-fragment rows = K).
// Layer-0 weights: register-resident B fragments (K=16 input is zero-padded).
// h state: double-buffered f16 in LDS, layout [row][k] for A-fragment loads.
// ---------------------------------------------------------------------------
__global__ __launch_bounds__(256) void lstm2_kernel(
    const float* __restrict__ x,
    const float* __restrict__ Wih0, const float* __restrict__ Whh0,
    const float* __restrict__ bih0, const float* __restrict__ bhh0,
    const float* __restrict__ Wih1, const float* __restrict__ Whh1,
    const float* __restrict__ bih1, const float* __restrict__ bhh1,
    float* __restrict__ hidden_out)
{
  __shared__ __align__(16) _Float16 WtI1[H][WSTR];       // [k][gatecol]
  __shared__ __align__(16) _Float16 WtH1[H][WSTR];
  __shared__ __align__(16) _Float16 h0buf[2][16][HSTR];  // [buf][row][k]
  __shared__ __align__(16) _Float16 h1buf[2][16][HSTR];
  __shared__ float bias0[G4];
  __shared__ float bias1[G4];

  const int  tid  = threadIdx.x;
  const int  wv   = tid >> 5;
  const int  lane = tid & 31;
  const int  mrow = lane & 15;
  const bool hiL  = lane >= 16;
  const int  k0   = hiL ? 8 : 0;
  const int  col  = wv * 16 + mrow;   // hidden column this (wave,lane) owns in C/D tiles

  // ---- stage layer-1 weights transposed into LDS (f32 -> f16) ----
  for (int idx = tid; idx < G4 * H; idx += 256) {
    const int g = idx / H, k = idx - g * H;
    WtI1[k][g] = (_Float16)Wih1[idx];
    WtH1[k][g] = (_Float16)Whh1[idx];
  }
  for (int i = tid; i < G4; i += 256) {
    bias0[i] = bih0[i] + bhh0[i];
    bias1[i] = bih1[i] + bhh1[i];
  }
  for (int idx = tid; idx < 2 * 16 * HSTR; idx += 256) {
    (&h0buf[0][0][0])[idx] = (_Float16)0.f;
    (&h1buf[0][0][0])[idx] = (_Float16)0.f;
  }

  // ---- layer-0 weights as register B fragments ----
  // B frag (32x16): lane L holds column nb+L%16, element e = K-row k0+(e<8?e:e+8).
  v16h bI0[4];
  v16h bH0[4][4];
  #pragma unroll
  for (int g = 0; g < 4; ++g) {
    const int nb = g * H + wv * 16;
    v16h f;
    #pragma unroll
    for (int e = 0; e < 16; ++e)
      f[e] = (e < 8) ? (_Float16)Wih0[(nb + mrow) * DF + k0 + e] : (_Float16)0.f;
    bI0[g] = f;
    #pragma unroll
    for (int kt = 0; kt < 4; ++kt) {
      const int k = kt * 32 + k0;
      v16h fh;
      #pragma unroll
      for (int e = 0; e < 16; ++e)
        fh[e] = (_Float16)Whh0[(nb + mrow) * H + k + (e < 8 ? e : e + 8)];
      bH0[kt][g] = fh;
    }
  }
  __syncthreads();

  const float bc0[4] = { bias0[0*H+col], bias0[1*H+col], bias0[2*H+col], bias0[3*H+col] };
  const float bc1[4] = { bias1[0*H+col], bias1[1*H+col], bias1[2*H+col], bias1[3*H+col] };

  const long nodebase = (long)blockIdx.x * 16;
  v8f c0, c1;
  #pragma unroll
  for (int e = 0; e < 8; ++e) { c0[e] = 0.f; c1[e] = 0.f; }

  int cur = 0;
  for (int t = 0; t < TT; ++t) {
    const int nxt = cur ^ 1;
    v8f acc[4];

    // ================= layer 0 =================
    #pragma unroll
    for (int g = 0; g < 4; ++g) {
      #pragma unroll
      for (int e = 0; e < 8; ++e) acc[g][e] = bc0[g];
    }
    {   // x_t contribution: K = 16 real, zero-padded to 32
      const float* xp = x + ((nodebase + mrow) * TT + t) * DF + k0;
      v16h ax;
      #pragma unroll
      for (int e = 0; e < 8; ++e)  ax[e] = (_Float16)xp[e];
      #pragma unroll
      for (int e = 8; e < 16; ++e) ax[e] = (_Float16)0.f;
      #pragma unroll
      for (int g = 0; g < 4; ++g) acc[g] = wmma_f16(ax, bI0[g], acc[g]);
    }
    #pragma unroll
    for (int kt = 0; kt < 4; ++kt) {   // h0 recurrence
      const _Float16* hp = &h0buf[cur][mrow][kt * 32 + k0];
      const v16h ah = cat16(*(const v8h*)hp, *(const v8h*)(hp + 16));
      #pragma unroll
      for (int g = 0; g < 4; ++g) acc[g] = wmma_f16(ah, bH0[kt][g], acc[g]);
    }
    #pragma unroll
    for (int e = 0; e < 8; ++e) {      // gates -> (c0, h0)
      const float iv = sigf(acc[0][e]);
      const float fv = sigf(acc[1][e]);
      const float gv = tanh_fast(acc[2][e]);
      const float ov = sigf(acc[3][e]);
      const float cv = fv * c0[e] + iv * gv;
      c0[e] = cv;
      h0buf[nxt][e + (hiL ? 8 : 0)][col] = (_Float16)(ov * tanh_fast(cv));
    }
    __syncthreads();

    // ================= layer 1 =================
    #pragma unroll
    for (int g = 0; g < 4; ++g) {
      #pragma unroll
      for (int e = 0; e < 8; ++e) acc[g][e] = bc1[g];
    }
    #pragma unroll
    for (int kt = 0; kt < 4; ++kt) {   // input = this step's layer-0 output
      const _Float16* hp = &h0buf[nxt][mrow][kt * 32 + k0];
      const v16h ah = cat16(*(const v8h*)hp, *(const v8h*)(hp + 16));
      const int r = kt * 32 + k0;
      #pragma unroll
      for (int g = 0; g < 4; ++g) {
        const _Float16* bp = &WtI1[r][g * H + col];
        v16h bw;
        #pragma unroll
        for (int e = 0; e < 16; ++e) bw[e] = bp[(e < 8 ? e : e + 8) * WSTR];
        acc[g] = wmma_f16(ah, bw, acc[g]);
      }
    }
    #pragma unroll
    for (int kt = 0; kt < 4; ++kt) {   // h1 recurrence
      const _Float16* hp = &h1buf[cur][mrow][kt * 32 + k0];
      const v16h ah = cat16(*(const v8h*)hp, *(const v8h*)(hp + 16));
      const int r = kt * 32 + k0;
      #pragma unroll
      for (int g = 0; g < 4; ++g) {
        const _Float16* bp = &WtH1[r][g * H + col];
        v16h bw;
        #pragma unroll
        for (int e = 0; e < 16; ++e) bw[e] = bp[(e < 8 ? e : e + 8) * WSTR];
        acc[g] = wmma_f16(ah, bw, acc[g]);
      }
    }
    #pragma unroll
    for (int e = 0; e < 8; ++e) {
      const float iv = sigf(acc[0][e]);
      const float fv = sigf(acc[1][e]);
      const float gv = tanh_fast(acc[2][e]);
      const float ov = sigf(acc[3][e]);
      const float cv = fv * c1[e] + iv * gv;
      c1[e] = cv;
      const float hv = ov * tanh_fast(cv);
      h1buf[nxt][e + (hiL ? 8 : 0)][col] = (_Float16)hv;
      if (t == TT - 1)
        hidden_out[(nodebase + e + (hiL ? 8 : 0)) * H + col] = hv;
    }
    __syncthreads();
    cur = nxt;
  }
}

// ---------------------------------------------------------------------------
// u-vector precompute: s_src = hidden . (W_t^T a[:H]) + b_t.a[:H], etc.
// u[0:128]=u_src, u[128:256]=u_dst, u[256]=c_src, u[257]=c_dst
// ---------------------------------------------------------------------------
__global__ __launch_bounds__(128) void uvec_kernel(
    const float* __restrict__ Wt, const float* __restrict__ bt,
    const float* __restrict__ a,  float* __restrict__ u)
{
  const int k = threadIdx.x;
  float s1 = 0.f, s2 = 0.f;
  for (int h = 0; h < H; ++h) {
    const float w = Wt[h * H + k];
    s1 += w * a[h];
    s2 += w * a[H + h];
  }
  u[k] = s1;
  u[H + k] = s2;
  if (k == 0) {
    float c1 = 0.f, c2 = 0.f;
    for (int h = 0; h < H; ++h) { c1 += bt[h] * a[h]; c2 += bt[h] * a[H + h]; }
    u[2 * H] = c1;
    u[2 * H + 1] = c2;
  }
}

// ---------------------------------------------------------------------------
// Per-node: s_src/s_dst dots + f16 copy of hidden. One wave per node.
// ---------------------------------------------------------------------------
__global__ __launch_bounds__(32) void srcdst_kernel(
    const float* __restrict__ hidden, const float* __restrict__ u,
    float* __restrict__ ssrc, float* __restrict__ sdst,
    _Float16* __restrict__ hidden_h)
{
  const int  i    = blockIdx.x;
  const int  lane = threadIdx.x;
  const float* hrow = hidden + (long)i * H;
  float a1 = 0.f, a2 = 0.f;
  #pragma unroll
  for (int k = lane; k < H; k += 32) {
    const float v = hrow[k];
    hidden_h[(long)i * H + k] = (_Float16)v;
    a1 += v * u[k];
    a2 += v * u[H + k];
  }
  #pragma unroll
  for (int off = 16; off > 0; off >>= 1) {
    a1 += __shfl_xor(a1, off, 32);
    a2 += __shfl_xor(a2, off, 32);
  }
  if (lane == 0) {
    ssrc[i] = a1 + u[2 * H];
    sdst[i] = a2 + u[2 * H + 1];
  }
}

// ---------------------------------------------------------------------------
// Flash-style dense graph attention: one wave owns 16 rows x full H=128.
// Adjacency tiles (16x32 f32 per wave) are streamed into LDS with
// GLOBAL_LOAD_ASYNC_TO_LDS_B128, double-buffered and synchronized with
// s_wait_asynccnt, so HBM latency overlaps the exp-heavy softmax VALU work
// and the P@V WMMAs. Row max/denom combined with shfl -- no barriers.
// ---------------------------------------------------------------------------
__global__ __launch_bounds__(64) void attn_kernel(
    const float* __restrict__ hidden_in, const _Float16* __restrict__ hidden_h,
    const float* __restrict__ adj,
    const float* __restrict__ ssrc, const float* __restrict__ sdst,
    float* __restrict__ hidden_out)
{
  // [wave][buf][row][col], col padded 32->36 (rows stay 16B aligned, fewer
  // bank conflicts on the strided consumer reads)
  __shared__ __align__(16) float adjt[2][2][16][36];

  const int  wv   = threadIdx.x >> 5;
  const int  lane = threadIdx.x & 31;
  const int  mrow = lane & 15;
  const bool hiL  = lane >= 16;
  const int  k0   = hiL ? 8 : 0;
  const int  coloff = hiL ? 16 : 0;   // which 16-float half of the tile row this lane fetches
  const long row0  = ((long)blockIdx.x * 2 + wv) * 16;
  const long myrow = row0 + mrow;

  const float  sd     = sdst[myrow];
  const float* adjrow = adj + myrow * (long)NNODES;

  float* ld0 = &adjt[wv][0][mrow][coloff];
  float* ld1 = &adjt[wv][1][mrow][coloff];

  float mrun = -1e30f, drun = 0.f;
  v8f acc[8];
  #pragma unroll
  for (int nt = 0; nt < 8; ++nt) {
    #pragma unroll
    for (int e = 0; e < 8; ++e) acc[nt][e] = 0.f;
  }

  // prologue: stream tile 0 into buffer 0 (4 x b128 per lane = 64B)
  {
    const float* g = adjrow + coloff;
    ASYNC_B128(g,      ld0);
    ASYNC_B128(g + 4,  ld0 + 4);
    ASYNC_B128(g + 8,  ld0 + 8);
    ASYNC_B128(g + 12, ld0 + 12);
  }

  for (int jt = 0; jt < NTILE; ++jt) {
    const int buf = jt & 1;
    if (jt + 1 < NTILE) {
      // stream next tile into the other buffer, then wait for current tile
      float* dst = buf ? ld0 : ld1;
      const float* g = adjrow + (jt + 1) * 32 + coloff;
      ASYNC_B128(g,      dst);
      ASYNC_B128(g + 4,  dst + 4);
      ASYNC_B128(g + 8,  dst + 8);
      ASYNC_B128(g + 12, dst + 12);
      WAIT_ASYNC(4);     // in-order completion: first 4 (tile jt) are done
    } else {
      WAIT_ASYNC(0);
    }

    const float* at = &adjt[wv][buf][mrow][0];
    const int jb = jt * 32 + k0;
    float p[16];
    #pragma unroll
    for (int e = 0; e < 8; ++e) {
      float s1 = sd + ssrc[jb + e];
      float s2 = sd + ssrc[jb + 16 + e];
      s1 = (s1 > 0.f) ? s1 : 0.01f * s1;
      s2 = (s2 > 0.f) ? s2 : 0.01f * s2;
      p[e]     = s1 * at[k0 + e];
      p[8 + e] = s2 * at[k0 + 16 + e];
    }
    float tmax = p[0];
    #pragma unroll
    for (int e = 1; e < 16; ++e) tmax = fmaxf(tmax, p[e]);
    tmax = fmaxf(tmax, __shfl_xor(tmax, 16, 32));        // full-row tile max
    const float newm  = fmaxf(mrun, tmax);
    const float scale = __expf(mrun - newm);             // 0 on first tile

    v16h pf;
    float tsum = 0.f;
    #pragma unroll
    for (int e = 0; e < 16; ++e) {
      const float ev = __expf(p[e] - newm);
      tsum += ev;
      pf[e] = (_Float16)ev;
    }
    tsum += __shfl_xor(tsum, 16, 32);
    drun = drun * scale + tsum;
    mrun = newm;

    float sc[8];                                          // per-C/D-row rescale
    #pragma unroll
    for (int e = 0; e < 8; ++e) sc[e] = __shfl(scale, e + (hiL ? 8 : 0), 32);

    const _Float16* hrow = hidden_h + (long)(jt * 32 + k0) * H + mrow;  // B frag: n = lane%16, K = k0+(e<8?e:e+8)
    #pragma unroll
    for (int nt = 0; nt < 8; ++nt) {
      v16h bf;
      #pragma unroll
      for (int e = 0; e < 16; ++e) bf[e] = hrow[(long)(e < 8 ? e : e + 8) * H + nt * 16];
      #pragma unroll
      for (int e = 0; e < 8; ++e) acc[nt][e] *= sc[e];
      acc[nt] = wmma_f16(pf, bf, acc[nt]);
    }
  }

  #pragma unroll
  for (int e = 0; e < 8; ++e) {
    const int   m  = e + (hiL ? 8 : 0);
    const float dd = __shfl(drun, m, 32);
    const long  orow = row0 + m;
    #pragma unroll
    for (int nt = 0; nt < 8; ++nt) {
      const int n = nt * 16 + mrow;
      hidden_out[orow * H + n] = acc[nt][e] / dd + hidden_in[orow * H + n];
    }
  }
}

// ---------------------------------------------------------------------------
// Final FC: out = leaky(hidden @ W_fc^T + b_fc) @ W_out^T + b_out  (C_OUT=1)
// ---------------------------------------------------------------------------
__global__ __launch_bounds__(32) void fc_kernel(
    const float* __restrict__ hidden,
    const float* __restrict__ Wfc, const float* __restrict__ bfc,
    const float* __restrict__ Wout, const float* __restrict__ bout,
    float* __restrict__ out)
{
  const int i    = blockIdx.x;
  const int lane = threadIdx.x;
  const float* hrow = hidden + (long)i * H;
  float s = 0.f;
  for (int h = lane; h < H; h += 32) {
    float a = bfc[h];
    for (int k = 0; k < H; ++k) a += Wfc[h * H + k] * hrow[k];
    a = (a > 0.f) ? a : 0.01f * a;
    s += Wout[h] * a;
  }
  #pragma unroll
  for (int off = 16; off > 0; off >>= 1) s += __shfl_xor(s, off, 32);
  if (lane == 0) out[i] = s + bout[0];
}

// ---------------------------------------------------------------------------
extern "C" void kernel_launch(void* const* d_in, const int* in_sizes, int n_in,
                              void* d_out, int out_size, void* d_ws, size_t ws_size,
                              hipStream_t stream)
{
  (void)in_sizes; (void)n_in; (void)out_size; (void)ws_size;
  const float* x    = (const float*)d_in[0];
  const float* adj  = (const float*)d_in[1];
  const float* Wih0 = (const float*)d_in[2];
  const float* Whh0 = (const float*)d_in[3];
  const float* bih0 = (const float*)d_in[4];
  const float* bhh0 = (const float*)d_in[5];
  const float* Wih1 = (const float*)d_in[6];
  const float* Whh1 = (const float*)d_in[7];
  const float* bih1 = (const float*)d_in[8];
  const float* bhh1 = (const float*)d_in[9];
  const float* Wtr  = (const float*)d_in[10];
  const float* btr  = (const float*)d_in[11];
  const float* av   = (const float*)d_in[12];
  const float* Wfc  = (const float*)d_in[13];
  const float* bfc  = (const float*)d_in[14];
  const float* Wout = (const float*)d_in[15];
  const float* bout = (const float*)d_in[16];

  // workspace carve-up (~5.2 MB)
  float*    hidA = (float*)d_ws;                         // N*H f32
  float*    hidB = hidA + (long)NNODES * H;              // N*H f32
  _Float16* hidh = (_Float16*)(hidB + (long)NNODES * H); // N*H f16
  float*    ssrc = (float*)(hidh + (long)NNODES * H);
  float*    sdst = ssrc + NNODES;
  float*    uvec = sdst + NNODES;                        // 2H+2 floats

  lstm2_kernel<<<NNODES / 16, 256, 0, stream>>>(
      x, Wih0, Whh0, bih0, bhh0, Wih1, Whh1, bih1, bhh1, hidA);

  uvec_kernel<<<1, 128, 0, stream>>>(Wtr, btr, av, uvec);

  // attention round 0: hidA -> hidB
  srcdst_kernel<<<NNODES, 32, 0, stream>>>(hidA, uvec, ssrc, sdst, hidh);
  attn_kernel<<<NNODES / 32, 64, 0, stream>>>(hidA, hidh, adj, ssrc, sdst, hidB);

  // attention round 1: hidB -> hidA
  srcdst_kernel<<<NNODES, 32, 0, stream>>>(hidB, uvec, ssrc, sdst, hidh);
  attn_kernel<<<NNODES / 32, 64, 0, stream>>>(hidB, hidh, adj, ssrc, sdst, hidA);

  fc_kernel<<<NNODES, 32, 0, stream>>>(hidA, Wfc, bfc, Wout, bout, (float*)d_out);
}
